// MeanCentroidClassifier_9620726743322
// MI455X (gfx1250) — matrix operand order repaired and checked by reading, not compiled, not run on hardware
//
#include <hip/hip_runtime.h>

// ---------------------------------------------------------------------------
// MI455X cosine-similarity classifier:
//   out = -(normalize(x) @ normalize(centroids)^T)
// Phase 1: row-normalize to f16 in workspace (negation folded into x's scale).
// Phase 2: f16 WMMA GEMM, f32 accumulate (v_wmma_f32_16x16x32_f16).
//          Wave tile 64x64 = 4x4 WMMA tiles: 16 b128 loads per 16 WMMAs,
//          512 B of L2 traffic per WMMA (everything is L2-resident; 192 MB).
// ---------------------------------------------------------------------------

typedef __attribute__((ext_vector_type(16))) _Float16 v16h;
typedef __attribute__((ext_vector_type(8)))  _Float16 v8h;
typedef __attribute__((ext_vector_type(4)))  _Float16 v4h;
typedef __attribute__((ext_vector_type(8)))  float    v8f;

#define DDIM 1024  // feature dimension (both inputs)

// One 256-thread block per row: sum-of-squares reduce, scale, f32 -> f16.
__global__ void rownorm_f16_kernel(const float* __restrict__ src,
                                   _Float16* __restrict__ dst,
                                   float sign)
{
    __shared__ float red[256];
    const int row = blockIdx.x;
    const int tid = threadIdx.x;

    const float4 v = ((const float4*)(src + (size_t)row * DDIM))[tid];
    float ss = v.x * v.x + v.y * v.y + v.z * v.z + v.w * v.w;
    red[tid] = ss;
    __syncthreads();
    #pragma unroll
    for (int off = 128; off > 0; off >>= 1) {
        if (tid < off) red[tid] += red[tid + off];
        __syncthreads();
    }
    // torch cosine_similarity clamps the norm at 1e-8
    const float scale = sign / fmaxf(sqrtf(red[0]), 1e-8f);

    v4h o;
    o[0] = (_Float16)(v.x * scale);
    o[1] = (_Float16)(v.y * scale);
    o[2] = (_Float16)(v.z * scale);
    o[3] = (_Float16)(v.w * scale);
    ((v4h*)(dst + (size_t)row * DDIM))[tid] = o;
}

// A-fragment (16-bit 16x32): lanes 0-15 hold K 0..15 of row M=lane;
// lanes 16-31 hold K 16..31 (same contiguous k order as B).
// => per lane: two contiguous 16B loads at k + 16*halfsel and k + 16*halfsel + 8.
__device__ __forceinline__ v16h load_a_frag(const _Float16* rowp, int k, int hl)
{
    v8h lo = *(const v8h*)(rowp + k + hl * 16);
    v8h hi = *(const v8h*)(rowp + k + hl * 16 + 8);
    return __builtin_shufflevector(lo, hi, 0, 1, 2, 3, 4, 5, 6, 7,
                                           8, 9, 10, 11, 12, 13, 14, 15);
}

// B-fragment (16-bit 32x16): lane n holds K=0..15 (lanes 0-15) or K=16..31
// (lanes 16-31) of column N=lane%16. Since B = cn^T, column N of B is row N of
// cn, contiguous along d => one 32B load per lane.
__device__ __forceinline__ v16h load_b_frag(const _Float16* rowp, int k, int hl)
{
    return *(const v16h*)(rowp + k + hl * 16);
}

// 8 waves/block arranged 4(M) x 2(N); block tile 256(M) x 128(N).
// Wave tile 64x64 = 4x4 grid of 16x16 WMMA tiles.
__global__ void __launch_bounds__(256)
cosgemm_wmma_kernel(const _Float16* __restrict__ A,   // -xn  [M x 1024]
                    const _Float16* __restrict__ B,   //  cn  [N x 1024]
                    float* __restrict__ out,          //      [M x N]
                    int Ncols)
{
    const int lane = threadIdx.x & 31;
    const int wave = threadIdx.x >> 5;   // 0..7
    const int l16  = lane & 15;
    const int hl   = lane >> 4;          // half-wave selector

    const int wm = wave >> 1;            // 4 waves along M
    const int wn = wave & 1;             // 2 waves along N

    const int mbase = blockIdx.y * 256 + wm * 64;
    const int nbase = blockIdx.x * 128 + wn * 64;

    const _Float16* arow[4];
    #pragma unroll
    for (int mt = 0; mt < 4; ++mt)
        arow[mt] = A + (size_t)(mbase + mt * 16 + l16) * DDIM;
    const _Float16* brow[4];
    #pragma unroll
    for (int nt = 0; nt < 4; ++nt)
        brow[nt] = B + (size_t)(nbase + nt * 16 + l16) * DDIM;

    v8f acc[4][4];
    #pragma unroll
    for (int mt = 0; mt < 4; ++mt)
        #pragma unroll
        for (int nt = 0; nt < 4; ++nt) {
            v8f z = {};
            acc[mt][nt] = z;
        }

    for (int k = 0; k < DDIM; k += 32) {
        v16h af[4], bf[4];
        #pragma unroll
        for (int mt = 0; mt < 4; ++mt) af[mt] = load_a_frag(arow[mt], k, hl);
        #pragma unroll
        for (int nt = 0; nt < 4; ++nt) bf[nt] = load_b_frag(brow[nt], k, hl);

        #pragma unroll
        for (int mt = 0; mt < 4; ++mt)
            #pragma unroll
            for (int nt = 0; nt < 4; ++nt)
                acc[mt][nt] = __builtin_amdgcn_wmma_f32_16x16x32_f16(
                    /*neg_a=*/false, af[mt], /*neg_b=*/false, bf[nt],
                    /*c_mod=*/(short)0, acc[mt][nt],
                    /*reuse_a=*/false, /*reuse_b=*/false);
    }

    // C/D layout: VGPR r, lanes 0-15 -> (M=r, N=lane); lanes 16-31 -> (M=8+r).
    #pragma unroll
    for (int mt = 0; mt < 4; ++mt)
        #pragma unroll
        for (int nt = 0; nt < 4; ++nt) {
            float* o = out + (size_t)(mbase + mt * 16 + hl * 8) * Ncols
                           + nbase + nt * 16 + l16;
            #pragma unroll
            for (int r = 0; r < 8; ++r)
                o[(size_t)r * Ncols] = acc[mt][nt][r];
        }
}

extern "C" void kernel_launch(void* const* d_in, const int* in_sizes, int n_in,
                              void* d_out, int out_size, void* d_ws, size_t ws_size,
                              hipStream_t stream)
{
    const float* x = (const float*)d_in[0];   // [Bx, 1024]
    const float* c = (const float*)d_in[1];   // [Nc, 1024]
    float* out = (float*)d_out;               // [Bx, Nc]

    const int Bx = in_sizes[0] / DDIM;        // 16384
    const int Nc = in_sizes[1] / DDIM;        // 1024

    _Float16* xh = (_Float16*)d_ws;           // 32 MB: -normalize(x) in f16
    _Float16* ch = xh + (size_t)Bx * DDIM;    //  2 MB:  normalize(c) in f16

    // Fold the output negation into x's scale: out = (-xn) . cn
    rownorm_f16_kernel<<<Bx, 256, 0, stream>>>(x, xh, -1.0f);
    rownorm_f16_kernel<<<Nc, 256, 0, stream>>>(c, ch, 1.0f);

    dim3 grid(Nc / 128, Bx / 256);            // 8 x 64 blocks
    cosgemm_wmma_kernel<<<grid, 256, 0, stream>>>(xh, ch, out, Nc);
}
